// RoPE_Attention_float_1975684956728
// MI455X (gfx1250) — matrix operand order repaired and checked by reading, compiled, not run on hardware
//
#include <hip/hip_runtime.h>

// Problem constants (match reference)
#define BB 4
#define SS 2048
#define DD 1024
#define KT 128              // keys per flash-attention chunk
#define NEG_BIG (-1.0e30f)
#define SM_SCALE (0.03125f) // 1/sqrt(1024)

typedef __bf16 bf16_t;
typedef __attribute__((ext_vector_type(16))) __bf16 v16bf;
typedef __attribute__((ext_vector_type(8)))  float  v8f;

__device__ __forceinline__ v8f v8f_zero() {
  v8f z = {0.f, 0.f, 0.f, 0.f, 0.f, 0.f, 0.f, 0.f};
  return z;
}

// ---- WMMA fragment loaders -------------------------------------------------
// A fragment (16x32 bf16). p must point at: rowBase + (lane>>4)*16 elements.
// lanes 0-15 take K {0..15}; lanes 16-31 take K {16..31}.
__device__ __forceinline__ v16bf a_frag_bf16(const bf16_t* p, int kk) {
  v16bf a;
#pragma unroll
  for (int i = 0; i < 8; ++i) { a[i] = p[kk + i]; a[8 + i] = p[kk + 8 + i]; }
  return a;
}
// B fragment (32x16 bf16). p must point at: colRowBase + (lane>>4)*16 elements
// (lane&15 selects the N column; elements are contiguous along K).
__device__ __forceinline__ v16bf b_frag_bf16(const bf16_t* __restrict__ p, int kk) {
  v16bf b;
#pragma unroll
  for (int i = 0; i < 16; ++i) b[i] = p[kk + i];
  return b;
}

__device__ __forceinline__ v8f wmma_bf16(v16bf a, v16bf b, v8f c) {
  return __builtin_amdgcn_wmma_f32_16x16x32_bf16(false, a, false, b, (short)0, c, false, false);
}

// ============================================================================
// Kernel 0: one-shot fp32 -> bf16 conversion of Wq/Wk/Wv.
// Grid: (D*D/8/256, 3) x 256.
// ============================================================================
__global__ __launch_bounds__(256) void cvt_weights_kernel(
    const float* __restrict__ Wq, const float* __restrict__ Wk,
    const float* __restrict__ Wv,
    bf16_t* __restrict__ Wqb, bf16_t* __restrict__ Wkb, bf16_t* __restrict__ Wvb)
{
  const float* src = (blockIdx.y == 0) ? Wq : (blockIdx.y == 1) ? Wk : Wv;
  bf16_t*      dst = (blockIdx.y == 0) ? Wqb : (blockIdx.y == 1) ? Wkb : Wvb;
  const size_t e = ((size_t)blockIdx.x * 256 + threadIdx.x) * 8;
#pragma unroll
  for (int i = 0; i < 8; ++i) dst[e + i] = (bf16_t)src[e + i];
}

// ============================================================================
// Kernel 1: Q/K/V projections + bias + learned RoPE (software-pipelined).
//   x tile (16 x 1024) staged once per block into LDS as bf16 (shared by all
//   8 waves); weights stream as bf16 B-fragments (32B contiguous per lane).
//   Q,K -> bf16 row-major [B*S, D];  V -> bf16 TRANSPOSED [B, D, S]
// Grid: (D/16/8, B*S/16) x 256 threads.
// ============================================================================
__global__ __launch_bounds__(256) void qkv_rope_kernel(
    const float* __restrict__ x,  const float* __restrict__ nc,
    const bf16_t* __restrict__ Wqb, const bf16_t* __restrict__ Wkb,
    const bf16_t* __restrict__ Wvb,
    const float* __restrict__ bq, const float* __restrict__ bk,
    const float* __restrict__ bv, const float* __restrict__ Wr,
    bf16_t* __restrict__ Qb, bf16_t* __restrict__ Kb, bf16_t* __restrict__ Vt)
{
  __shared__ bf16_t xs[16][DD + 8];   // +8 bf16 pad -> rows skewed by 4 banks

  const int lane = threadIdx.x & 31;
  const int wave = threadIdx.x >> 5;
  const int r    = lane & 15;
  const int hi   = lane >> 4;

  const int tileN = blockIdx.x * 8 + wave;   // 0..63
  const int tileM = blockIdx.y;              // 0..511
  const int rowM  = tileM * 16;
  const int ncol  = tileN * 16 + r;          // lane's output column (B/C frag)

  // ---- stage + convert the shared x tile (16 x 1024 fp32 -> bf16) ---------
#pragma unroll
  for (int it = 0; it < 8; ++it) {
    const int e   = (threadIdx.x + it * 256) * 8;
    const int row = e >> 10;
    const int col = e & 1023;
    const float* s = x + (size_t)(rowM + row) * DD + col;
#pragma unroll
    for (int i = 0; i < 8; ++i) xs[row][col + i] = (bf16_t)s[i];
  }
  __syncthreads();

  const bf16_t* xrow = &xs[r][hi * 16];
  const bf16_t* wqr  = Wqb + (size_t)ncol * DD + hi * 16;
  const bf16_t* wkr  = Wkb + (size_t)ncol * DD + hi * 16;
  const bf16_t* wvr  = Wvb + (size_t)ncol * DD + hi * 16;

  v8f aq = v8f_zero(), ak = v8f_zero(), av = v8f_zero();

  // Pipelined K-loop: fragments for step kk+32 issued before WMMAs on kk.
  v16bf a  = a_frag_bf16(xrow, 0);
  v16bf fq = b_frag_bf16(wqr, 0);
  v16bf fk = b_frag_bf16(wkr, 0);
  v16bf fv = b_frag_bf16(wvr, 0);
#pragma unroll 4
  for (int kk = 0; kk < DD - 32; kk += 32) {
    v16bf a2 = a_frag_bf16(xrow, kk + 32);
    v16bf q2 = b_frag_bf16(wqr,  kk + 32);
    v16bf k2 = b_frag_bf16(wkr,  kk + 32);
    v16bf v2 = b_frag_bf16(wvr,  kk + 32);
    aq = wmma_bf16(a, fq, aq);
    ak = wmma_bf16(a, fk, ak);
    av = wmma_bf16(a, fv, av);
    a = a2; fq = q2; fk = k2; fv = v2;
  }
  aq = wmma_bf16(a, fq, aq);
  ak = wmma_bf16(a, fk, ak);
  av = wmma_bf16(a, fv, av);

  const float biq = bq[ncol], bik = bk[ncol], biv = bv[ncol];
  // learned RoPE: freq_i = nc0*Wr[i,0] + nc1*Wr[i,1]; pair (2i, 2i+1) rotated.
  const int   i2   = ncol >> 1;
  const float wr0  = Wr[2 * i2], wr1 = Wr[2 * i2 + 1];
  const bool  ev   = ((ncol & 1) == 0);
  const int   bidx = rowM / SS;              // tile never straddles a batch

#pragma unroll
  for (int v = 0; v < 8; ++v) {
    const int   grow = rowM + v + 8 * hi;    // C-frag M = v + 8*hi
    const float q  = aq[v] + biq;
    const float k  = ak[v] + bik;
    const float vv = av[v] + biv;
    const float n0 = nc[2 * grow], n1 = nc[2 * grow + 1];
    const float fr = n0 * wr0 + n1 * wr1;
    float sn, cs;
    __sincosf(fr, &sn, &cs);
    // pair partner lives in the adjacent lane (same freq since i2 matches)
    const float qp = __shfl_xor(q, 1, 32);
    const float kp = __shfl_xor(k, 1, 32);
    const float qo = ev ? (q * cs - qp * sn) : (qp * sn + q * cs);
    const float ko = ev ? (k * cs - kp * sn) : (kp * sn + k * cs);
    Qb[(size_t)grow * DD + ncol] = (bf16_t)qo;
    Kb[(size_t)grow * DD + ncol] = (bf16_t)ko;
    const int srow = grow - bidx * SS;
    Vt[(size_t)bidx * DD * SS + (size_t)ncol * SS + srow] = (bf16_t)vv;
  }
}

// ============================================================================
// Kernel 2: causal flash attention (software-pipelined WMMA streams).
// Block = (batch, 16-query tile), 8 waves; wave w owns D-slice [w*128, +128).
// ============================================================================
__global__ __launch_bounds__(256) void attn_kernel(
    const bf16_t* __restrict__ Qb, const bf16_t* __restrict__ Kb,
    const bf16_t* __restrict__ Vt, float* __restrict__ out)
{
  __shared__ bf16_t Qs[16][DD + 8];    // 33 KB staged Q tile
  __shared__ float  Ssc[16][KT + 4];   // scores (fp32)
  __shared__ bf16_t Pp[16][KT + 8];    // probabilities (bf16)
  __shared__ float  mRow[16], lRow[16], alphaRow[16], invRow[16];

  const int lane = threadIdx.x & 31;
  const int wave = threadIdx.x >> 5;
  const int r    = lane & 15;
  const int hi   = lane >> 4;

  const int b     = blockIdx.x >> 7;
  const int qt    = blockIdx.x & 127;
  const int qbase = qt * 16;
  const int qhi   = qbase + 15;

  const bf16_t* Vbat = Vt + (size_t)b * DD * SS;

  // ---- stage Q tile (16 x 1024 bf16) --------------------------------------
#pragma unroll
  for (int it = 0; it < 8; ++it) {
    const int e   = (threadIdx.x + it * 256) * 8;
    const int row = e >> 10;
    const int col = e & 1023;
    const bf16_t* s = Qb + (size_t)(b * SS + qbase + row) * DD + col;
#pragma unroll
    for (int i = 0; i < 8; ++i) Qs[row][col + i] = s[i];
  }
  if (threadIdx.x < 16) { mRow[threadIdx.x] = NEG_BIG; lRow[threadIdx.x] = 0.f; }
  v8f acc[8];
#pragma unroll
  for (int t = 0; t < 8; ++t) acc[t] = v8f_zero();
  __syncthreads();

  const bf16_t* Qrow  = &Qs[r][hi * 16];    // A-frag base (LDS)
  const int     dbase = wave * 128;

  const int nchunks = qhi / KT + 1;        // causal: skip fully-masked chunks
  for (int c = 0; c < nchunks; ++c) {
    const int kbase = c * KT;

    // ---- Phase A: score tile S = (Q K^T) * scale (pipelined) --------------
    {
      const int kcol = kbase + wave * 16 + r;                 // key idx (B col)
      const bf16_t* Krow = Kb + (size_t)(b * SS + kcol) * DD + hi * 16;
      v8f s = v8f_zero();
      v16bf a  = a_frag_bf16(Qrow, 0);
      v16bf bb = b_frag_bf16(Krow, 0);
#pragma unroll 4
      for (int kk = 0; kk < DD - 32; kk += 32) {
        __builtin_prefetch(Krow + kk + 256, 0, 1);            // global_prefetch_b8
        v16bf a2 = a_frag_bf16(Qrow, kk + 32);
        v16bf b2 = b_frag_bf16(Krow, kk + 32);
        s = wmma_bf16(a, bb, s);
        a = a2; bb = b2;
      }
      s = wmma_bf16(a, bb, s);
#pragma unroll
      for (int v = 0; v < 8; ++v)
        Ssc[v + 8 * hi][wave * 16 + r] = s[v] * SM_SCALE;
    }
    __syncthreads();

    // ---- Phase B: online softmax (one thread per query row) ---------------
    if (threadIdx.x < 16) {
      const int row  = threadIdx.x;
      const int qrow = qbase + row;
      const int jmax = min(KT, qrow - kbase + 1);             // causal bound (>=1)
      const float m0 = mRow[row];
      float mx = m0;
      for (int j = 0; j < jmax; ++j) mx = fmaxf(mx, Ssc[row][j]);
      const float al = __expf(m0 - mx);
      float sum = 0.f;
      for (int j = 0; j < KT; ++j) {
        const float p = (j < jmax) ? __expf(Ssc[row][j] - mx) : 0.f;
        Pp[row][j] = (bf16_t)p;
        sum += p;
      }
      mRow[row]     = mx;
      lRow[row]     = al * lRow[row] + sum;
      alphaRow[row] = al;
    }
    __syncthreads();

    // ---- Phase C: acc = acc*alpha + P @ V_chunk (pipelined 32-MMA chain) --
    {
      float alv[8];
#pragma unroll
      for (int v = 0; v < 8; ++v) alv[v] = alphaRow[v + 8 * hi];
#pragma unroll
      for (int t = 0; t < 8; ++t)
#pragma unroll
        for (int v = 0; v < 8; ++v) acc[t][v] *= alv[v];

      v16bf afr[4];                                           // P A-frags (LDS)
      const bf16_t* Prow = &Pp[r][hi * 16];
#pragma unroll
      for (int ks = 0; ks < 4; ++ks) afr[ks] = a_frag_bf16(Prow, 32 * ks);

      // flattened (t,ks) chain; V B-fragment loaded one step ahead
      auto vp = [&](int idx) -> const bf16_t* {
        const int t = idx >> 2, ks = idx & 3;
        return Vbat + (size_t)(dbase + t * 16 + r) * SS + kbase + ks * 32 + hi * 16;
      };
      v16bf bcur = b_frag_bf16(vp(0), 0);
#pragma unroll
      for (int idx = 0; idx < 32; ++idx) {
        v16bf bnext = bcur;
        if (idx < 31) bnext = b_frag_bf16(vp(idx + 1), 0);
        acc[idx >> 2] = wmma_bf16(afr[idx & 3], bcur, acc[idx >> 2]);
        bcur = bnext;
      }
    }
    __syncthreads();
  }

  // ---- Epilogue: normalize by l and store fp32 ----------------------------
  if (threadIdx.x < 16) invRow[threadIdx.x] = 1.0f / lRow[threadIdx.x];
  __syncthreads();
  float inv[8];
#pragma unroll
  for (int v = 0; v < 8; ++v) inv[v] = invRow[v + 8 * hi];
  float* obase = out + (size_t)(b * SS + qbase) * DD;
#pragma unroll
  for (int t = 0; t < 8; ++t) {
    const int dcol = dbase + t * 16 + r;
#pragma unroll
    for (int v = 0; v < 8; ++v)
      obase[(size_t)(v + 8 * hi) * DD + dcol] = acc[t][v] * inv[v];
  }
}

// ============================================================================
extern "C" void kernel_launch(void* const* d_in, const int* in_sizes, int n_in,
                              void* d_out, int out_size, void* d_ws, size_t ws_size,
                              hipStream_t stream)
{
  (void)in_sizes; (void)n_in; (void)out_size; (void)ws_size;
  // inputs: x, norm_coord, causal_mask, batch_mask, Wq, bq, Wk, bk, Wv, bv, Wr
  const float* x  = (const float*)d_in[0];
  const float* nc = (const float*)d_in[1];
  const float* Wq = (const float*)d_in[4];
  const float* bq = (const float*)d_in[5];
  const float* Wk = (const float*)d_in[6];
  const float* bk = (const float*)d_in[7];
  const float* Wv = (const float*)d_in[8];
  const float* bv = (const float*)d_in[9];
  const float* Wr = (const float*)d_in[10];
  float* out = (float*)d_out;

  const size_t n  = (size_t)BB * SS * DD;   // 8,388,608 elements
  const size_t nw = (size_t)DD * DD;        // 1,048,576 elements
  bf16_t* Qb  = (bf16_t*)d_ws;              // 16 MB
  bf16_t* Kb  = Qb + n;                     // 16 MB
  bf16_t* Vt  = Kb + n;                     // 16 MB (transposed [B,D,S])
  bf16_t* Wqb = Vt + n;                     // 2 MB
  bf16_t* Wkb = Wqb + nw;                   // 2 MB
  bf16_t* Wvb = Wkb + nw;                   // 2 MB

  dim3 g0(nw / 8 / 256, 3);                 // (512, 3)
  cvt_weights_kernel<<<g0, 256, 0, stream>>>(Wq, Wk, Wv, Wqb, Wkb, Wvb);

  dim3 g1(DD / 16 / 8, (BB * SS) / 16);     // (8, 512), 8 waves/block
  qkv_rope_kernel<<<g1, 256, 0, stream>>>(x, nc, Wqb, Wkb, Wvb,
                                          bq, bk, bv, Wr, Qb, Kb, Vt);

  dim3 g2(BB * (SS / 16));                  // 512 blocks
  attn_kernel<<<g2, 256, 0, stream>>>(Qb, Kb, Vt, out);
}
